// SimplifiedT3Model_26422638805102
// MI455X (gfx1250) — matrix operand order repaired and checked by reading, not compiled, not run on hardware
//
#include <hip/hip_runtime.h>
#include <stdint.h>

// ---- model dims ----
#define cB  16
#define cS  256
#define cT  2048
#define cD  512
#define cH  8
#define cL  6
#define cFF 2048
#define cNM 80
#define cDH 64

// ---- WMMA types (CDNA5 gfx1250, wave32) ----
typedef __attribute__((ext_vector_type(16))) __bf16 v16bf;
typedef __attribute__((ext_vector_type(8)))  float  v8f;

union FragU { v16bf v; uint4 q[2]; };

__device__ __forceinline__ unsigned short bf16_rne(float f) {
  union { float f; unsigned int u; } c; c.f = f;
  unsigned int u = c.u;
  unsigned int r = u + 0x7fffu + ((u >> 16) & 1u);
  if ((u & 0x7f800000u) == 0x7f800000u) r = u;  // Inf/NaN passthrough
  return (unsigned short)(r >> 16);
}
__device__ __forceinline__ float bf16_f(unsigned short h) {
  union { float f; unsigned int u; } c; c.u = ((unsigned int)h) << 16;
  return c.f;
}
__device__ __forceinline__ v8f wmma_bf16(v16bf a, v16bf b, v8f c) {
  return __builtin_amdgcn_wmma_f32_16x16x32_bf16(false, a, false, b, (short)0, c, false, false);
}

// gfx1250 async global->LDS copy (ASYNCcnt-tracked, no VGPR round trip)
__device__ __forceinline__ void async_copy16(void* lds_dst, const void* gsrc) {
  asm volatile("global_load_async_to_lds_b128 %0, %1, off"
               :: "v"((unsigned int)(uintptr_t)lds_dst), "v"(gsrc)
               : "memory");
}
__device__ __forceinline__ void wait_async() {
  asm volatile("s_wait_asynccnt 0x0" ::: "memory");
}

#define FL_BIAS  1
#define FL_RELU  2
#define FL_OBF16 4

#define LSTR 40   // LDS row stride (ushorts): 32 + 8 pad, keeps 16B alignment

// ============================================================================
// Batched WMMA GEMM (all-NT): C = epilogue(A[MxK] * B[NxK]^T). bf16 in, f32 acc.
// Block 256 thr (8 waves), tile 128x64, wave -> 32x32 (4 wmma / chunk).
// LDS double-buffered; ALL staging via async global->LDS b128 copies which
// overlap the WMMAs on the previous chunk.
// Requires M%128==0, N%64==0, K%32==0 (true for all call sites).
// ============================================================================
__global__ __launch_bounds__(256) void k_gemm(
    const unsigned short* __restrict__ A, const unsigned short* __restrict__ Bm,
    const float* __restrict__ bias, void* __restrict__ Cout,
    int M, int N, int K, long long sA, long long sB, long long sC, int flags)
{
  __shared__ __align__(16) unsigned short As[2][128 * LSTR];
  __shared__ __align__(16) unsigned short Bs[2][64 * LSTR];
  const int tid  = threadIdx.x;
  const int lane = tid & 31, wv = tid >> 5;
  const int rw = wv & 3;        // 32-row block
  const int cw = wv >> 2;       // 32-col block
  const int bM0 = blockIdx.y * 128;
  const int bN0 = blockIdx.x * 64;
  A  += (long long)blockIdx.z * sA;
  Bm += (long long)blockIdx.z * sB;

  // ---- staging addresses ----
  const int arow = tid >> 1;              // 0..127
  const int akc  = (tid & 1) * 16;        // 0 or 16
  const unsigned short* gA = A + (long long)(bM0 + arow) * K + akc;
  unsigned short* ldsA[2] = { &As[0][arow * LSTR + akc], &As[1][arow * LSTR + akc] };

  const int brow = tid >> 2;              // 0..63
  const int bkc  = (tid & 3) * 8;
  const unsigned short* gB = Bm + (long long)(bN0 + brow) * K + bkc;
  unsigned short* ldsB[2] = { &Bs[0][brow * LSTR + bkc], &Bs[1][brow * LSTR + bkc] };

  // ---- prologue: stage chunk 0 into buffer 0 ----
  async_copy16(ldsA[0], gA);
  async_copy16(ldsA[0] + 8, gA + 8);
  async_copy16(ldsB[0], gB);
  wait_async();
  __syncthreads();

  v8f acc00 = {}, acc01 = {}, acc10 = {}, acc11 = {};
  const int m   = lane & 15;
  const int khA = (lane >> 4) * 16;   // A frag: k = khA + e (contiguous 16)
  const int khB = (lane >> 4) * 16;   // B frag: k = khB + e (contiguous 16)

  const int nsteps = K >> 5;
  for (int it = 0; it < nsteps; ++it) {
    const int cur = it & 1, nxt = cur ^ 1;
    const bool more = (it + 1 < nsteps);
    if (more) { // issue async copies for next chunk (overlap wmma below)
      const unsigned short* gAn = gA + (it + 1) * 32;
      async_copy16(ldsA[nxt], gAn);
      async_copy16(ldsA[nxt] + 8, gAn + 8);
      async_copy16(ldsB[nxt], gB + (it + 1) * 32);
    }
    const unsigned short* Ac = As[cur];
    const unsigned short* Bc = Bs[cur];
    FragU a0, a1, b0, b1;
    a0.q[0] = *(const uint4*)&Ac[(rw * 32 + m) * LSTR + khA];
    a0.q[1] = *(const uint4*)&Ac[(rw * 32 + m) * LSTR + khA + 8];
    a1.q[0] = *(const uint4*)&Ac[(rw * 32 + 16 + m) * LSTR + khA];
    a1.q[1] = *(const uint4*)&Ac[(rw * 32 + 16 + m) * LSTR + khA + 8];
    int nrow = cw * 32 + m;
    b0.q[0] = *(const uint4*)&Bc[nrow * LSTR + khB];
    b0.q[1] = *(const uint4*)&Bc[nrow * LSTR + khB + 8];
    b1.q[0] = *(const uint4*)&Bc[(nrow + 16) * LSTR + khB];
    b1.q[1] = *(const uint4*)&Bc[(nrow + 16) * LSTR + khB + 8];
    acc00 = wmma_bf16(a0.v, b0.v, acc00);
    acc01 = wmma_bf16(a0.v, b1.v, acc01);
    acc10 = wmma_bf16(a1.v, b0.v, acc10);
    acc11 = wmma_bf16(a1.v, b1.v, acc11);
    if (more) {
      wait_async();
      __syncthreads();
    }
  }

  // ---- epilogue ----
  const int nlo = lane & 15;
  const int mhi = (lane >> 4) * 8;
  float*          Cf = (float*)Cout          + (long long)blockIdx.z * sC;
  unsigned short* Cb = (unsigned short*)Cout + (long long)blockIdx.z * sC;
  v8f accs[4] = { acc00, acc01, acc10, acc11 };
  #pragma unroll
  for (int ti = 0; ti < 2; ++ti) {
    #pragma unroll
    for (int tj = 0; tj < 2; ++tj) {
      v8f acc = accs[ti * 2 + tj];
      #pragma unroll
      for (int r = 0; r < 8; ++r) {
        int row = bM0 + rw * 32 + ti * 16 + mhi + r;
        int col = bN0 + cw * 32 + tj * 16 + nlo;
        float v = acc[r];
        if (flags & FL_BIAS) v += bias[col];
        if (flags & FL_RELU) v = fmaxf(v, 0.f);
        long long o = (long long)row * N + col;
        if (flags & FL_OBF16) Cb[o] = bf16_rne(v);
        else                  Cf[o] = v;
      }
    }
  }
}

// ============================================================================
// Transpose-convert: dst[N x K] bf16 = transpose(src[K x N] f32).
// LDS-tiled 32x32, both sides coalesced. Guards handle N=80 (mel_W2).
// ============================================================================
__global__ __launch_bounds__(256) void k_convT(const float* __restrict__ src,
    unsigned short* __restrict__ dst, int K, int N) {
  __shared__ float tile[32][33];
  int n0 = blockIdx.x * 32;
  int k0 = blockIdx.y * 32;
  int tx = threadIdx.x & 31;
  int ty = threadIdx.x >> 5;          // 0..7
  #pragma unroll
  for (int j = 0; j < 4; ++j) {
    int k = k0 + ty + j * 8;
    int n = n0 + tx;
    tile[ty + j * 8][tx] = (k < K && n < N) ? src[(long long)k * N + n] : 0.f;
  }
  __syncthreads();
  #pragma unroll
  for (int j = 0; j < 4; ++j) {
    int n = n0 + ty + j * 8;
    int k = k0 + tx;
    if (n < N && k < K) dst[(long long)n * K + k] = bf16_rne(tile[tx][ty + j * 8]);
  }
}

// ============================================================================
// Elementwise / reduction kernels
// ============================================================================
__global__ void k_embed(const int* __restrict__ ids, const float* __restrict__ emb,
                        const float* __restrict__ pe, float* __restrict__ xf,
                        unsigned short* __restrict__ xb) {
  int i = blockIdx.x * 256 + threadIdx.x;           // over B*S*D
  if (i >= cB * cS * cD) return;
  int d = i % cD; int bs = i / cD; int s = bs % cS;
  float v = emb[(long long)ids[bs] * cD + d] + pe[(long long)s * cD + d];
  xf[i] = v; xb[i] = bf16_rne(v);
}

// Q,K -> [B,H,S,DH]; V -> transposed [B,H,DH,S] so attn*V runs as NT GEMM.
__global__ void k_split_qkv(const float* __restrict__ qkv, unsigned short* __restrict__ q,
                            unsigned short* __restrict__ k, unsigned short* __restrict__ vT) {
  int i = blockIdx.x * 256 + threadIdx.x;           // over B*S*D
  if (i >= cB * cS * cD) return;
  int d = i % cD; int bs = i / cD; int s = bs % cS; int b = bs / cS;
  int h = d / cDH, dh = d % cDH;
  long long dst  = (((long long)(b * cH + h)) * cS + s) * cDH + dh;
  long long dstT = (((long long)(b * cH + h)) * cDH + dh) * cS + s;
  const float* base = qkv + (long long)bs * (3 * cD);
  q[dst]   = bf16_rne(base[d]);
  k[dst]   = bf16_rne(base[cD + d]);
  vT[dstT] = bf16_rne(base[2 * cD + d]);
}

__global__ void k_merge_heads(const float* __restrict__ o, unsigned short* __restrict__ om) {
  int i = blockIdx.x * 256 + threadIdx.x;           // over B*S*D
  if (i >= cB * cS * cD) return;
  int d = i % cD; int bs = i / cD; int s = bs % cS; int b = bs / cS;
  int h = d / cDH, dh = d % cDH;
  om[i] = bf16_rne(o[(((long long)(b * cH + h)) * cS + s) * cDH + dh]);
}

__global__ __launch_bounds__(256) void k_softmax(const float* __restrict__ sc,
    unsigned short* __restrict__ attn, const int* __restrict__ lens, float scale) {
  int row = blockIdx.x;                              // 0 .. B*H*S-1
  int b = row / (cH * cS);
  int t = threadIdx.x;                               // key index (S=256)
  float v = sc[(long long)row * cS + t] * scale;
  if (t >= lens[b]) v = -1e9f;
  __shared__ float red[256];
  red[t] = v; __syncthreads();
  for (int s = 128; s > 0; s >>= 1) { if (t < s) red[t] = fmaxf(red[t], red[t + s]); __syncthreads(); }
  float mx = red[0]; __syncthreads();
  float e = __expf(v - mx);
  red[t] = e; __syncthreads();
  for (int s = 128; s > 0; s >>= 1) { if (t < s) red[t] += red[t + s]; __syncthreads(); }
  float inv = 1.f / red[0];
  attn[(long long)row * cS + t] = bf16_rne(e * inv);
}

__global__ __launch_bounds__(256) void k_add_ln(float* __restrict__ x, const float* __restrict__ y,
    const float* __restrict__ g, const float* __restrict__ bb, unsigned short* __restrict__ xb) {
  int row = blockIdx.x; int t = threadIdx.x;
  const long long off = (long long)row * cD;
  float a0 = x[off + t]       + y[off + t];
  float a1 = x[off + t + 256] + y[off + t + 256];
  __shared__ float red[256];
  red[t] = a0 + a1; __syncthreads();
  for (int s = 128; s > 0; s >>= 1) { if (t < s) red[t] += red[t + s]; __syncthreads(); }
  float mean = red[0] * (1.0f / cD); __syncthreads();
  float d0 = a0 - mean, d1 = a1 - mean;
  red[t] = d0 * d0 + d1 * d1; __syncthreads();
  for (int s = 128; s > 0; s >>= 1) { if (t < s) red[t] += red[t + s]; __syncthreads(); }
  float rstd = rsqrtf(red[0] * (1.0f / cD) + 1e-5f);
  float o0 = d0 * rstd * g[t]       + bb[t];
  float o1 = d1 * rstd * g[t + 256] + bb[t + 256];
  x[off + t] = o0;        x[off + t + 256] = o1;
  xb[off + t] = bf16_rne(o0); xb[off + t + 256] = bf16_rne(o1);
}

__global__ __launch_bounds__(256) void k_dur2(const unsigned short* __restrict__ h,
    const float* __restrict__ w2, const float* __restrict__ b2, float* __restrict__ dur) {
  int row = blockIdx.x; int t = threadIdx.x;         // K=256
  float p = bf16_f(h[(long long)row * 256 + t]) * w2[t];
  __shared__ float red[256];
  red[t] = p; __syncthreads();
  for (int s = 128; s > 0; s >>= 1) { if (t < s) red[t] += red[t + s]; __syncthreads(); }
  if (t == 0) {
    float x = red[0] + b2[0];
    dur[row] = (x > 20.f) ? x : log1pf(__expf(x));   // softplus
  }
}

__global__ __launch_bounds__(256) void k_cumsum(const float* __restrict__ dur, int* __restrict__ cum) {
  int b = blockIdx.x; int t = threadIdx.x;           // S=256
  __shared__ int sh[256];
  int dv = (int)rintf(dur[(long long)b * cS + t]);   // round-half-even like jnp.round
  if (dv < 1) dv = 1;
  sh[t] = dv; __syncthreads();
  for (int off = 1; off < 256; off <<= 1) {
    int add = (t >= off) ? sh[t - off] : 0; __syncthreads();
    sh[t] += add; __syncthreads();
  }
  cum[b * cS + t] = sh[t];
}

__global__ void k_idx(const int* __restrict__ cum, int* __restrict__ idx) {
  int i = blockIdx.x * 256 + threadIdx.x;            // B*T
  if (i >= cB * cT) return;
  int b = i / cT, t = i % cT;
  const int* c = cum + b * cS;
  int total = c[cS - 1];
  int lo = 0, hi = cS;                               // first s with c[s] > t
  while (lo < hi) { int mid = (lo + hi) >> 1; if (c[mid] <= t) lo = mid + 1; else hi = mid; }
  int id = lo; if (id > cS - 1) id = cS - 1;
  idx[i] = (t < total) ? id : -1;
}

__global__ void k_expand(const float* __restrict__ enc, const int* __restrict__ idx,
                         unsigned short* __restrict__ out) {
  long long i = (long long)blockIdx.x * 256 + threadIdx.x;   // B*T*D
  if (i >= (long long)cB * cT * cD) return;
  int d = (int)(i % cD); long long bt = i / cD;
  int b = (int)(bt / cT);
  int id = idx[bt];
  float v = (id >= 0) ? enc[((long long)b * cS + id) * cD + d] : 0.f;
  out[i] = bf16_rne(v);
}

// ============================================================================
// Fused mel decoder: mel[b,n,t] = relu(X[row]@W1 + b1) @ W2 + b2, transposed out.
// W1T is [FF x D], W2T is [NM x FF] (pre-transposed bf16), so ALL tile staging
// is async b128 copies. Block = 128 threads (4 waves), 64 rows, 80 cols.
// ============================================================================
__global__ __launch_bounds__(128) void k_mel(const unsigned short* __restrict__ X,
    const unsigned short* __restrict__ W1T, const float* __restrict__ b1,
    const unsigned short* __restrict__ W2T, const float* __restrict__ b2,
    float* __restrict__ mel)
{
  __shared__ __align__(16) unsigned short Xs[64 * LSTR];
  __shared__ __align__(16) unsigned short W1t[32 * LSTR];
  __shared__ __align__(16) unsigned short Hs[64 * LSTR];
  __shared__ __align__(16) unsigned short W2t[80 * LSTR];
  const int tid = threadIdx.x, lane = tid & 31, wv = tid >> 5;  // 4 waves
  const int r0 = blockIdx.x * 64;
  const int m   = lane & 15;
  const int khA = (lane >> 4) * 16;
  const int khB = (lane >> 4) * 16;
  const int nlo = lane & 15;
  const int mhi = (lane >> 4) * 8;

  const int xrow = tid >> 1;              // 0..63
  const int xkc  = (tid & 1) * 16;
  unsigned short* ldsX = &Xs[xrow * LSTR + xkc];
  const unsigned short* gX = X + (long long)(r0 + xrow) * cD + xkc;

  const int w1n  = tid >> 2;              // 0..31
  const int w1kc = (tid & 3) * 8;
  unsigned short* ldsW1 = &W1t[w1n * LSTR + w1kc];

  v8f out[5] = {};

  for (int ffc = 0; ffc < cFF; ffc += 32) {
    v8f h0 = {}; v8f h1 = {};
    for (int k0 = 0; k0 < cD; k0 += 32) {
      // X tile 64x32 and W1T tile 32x32, both via async copies
      async_copy16(ldsX, gX + k0);
      async_copy16(ldsX + 8, gX + k0 + 8);
      async_copy16(ldsW1, W1T + (long long)(ffc + w1n) * cD + k0 + w1kc);
      wait_async();
      __syncthreads();
      FragU a, f0, f1;
      a.q[0] = *(const uint4*)&Xs[(wv * 16 + m) * LSTR + khA];
      a.q[1] = *(const uint4*)&Xs[(wv * 16 + m) * LSTR + khA + 8];
      f0.q[0] = *(const uint4*)&W1t[m * LSTR + khB];
      f0.q[1] = *(const uint4*)&W1t[m * LSTR + khB + 8];
      f1.q[0] = *(const uint4*)&W1t[(m + 16) * LSTR + khB];
      f1.q[1] = *(const uint4*)&W1t[(m + 16) * LSTR + khB + 8];
      h0 = wmma_bf16(a.v, f0.v, h0);
      h1 = wmma_bf16(a.v, f1.v, h1);
      __syncthreads();
    }
    // bias + relu -> Hs (each wave writes its own 16-row strip)
    #pragma unroll
    for (int r = 0; r < 8; ++r) {
      int mm = mhi + r;
      float v0 = fmaxf(h0[r] + b1[ffc + nlo],      0.f);
      float v1 = fmaxf(h1[r] + b1[ffc + 16 + nlo], 0.f);
      Hs[(wv * 16 + mm) * LSTR + nlo]      = bf16_rne(v0);
      Hs[(wv * 16 + mm) * LSTR + 16 + nlo] = bf16_rne(v1);
    }
    // W2T tile [80n x 32k] via async copies
    for (int e = tid; e < 80 * 4; e += 128) {
      int n  = e >> 2;
      int kc = (e & 3) * 8;
      async_copy16(&W2t[n * LSTR + kc], W2T + (long long)n * cFF + ffc + kc);
    }
    wait_async();
    __syncthreads();
    { // stage 2: out[j] += H(16x32) * W2chunk(32x16)
      FragU a;
      a.q[0] = *(const uint4*)&Hs[(wv * 16 + m) * LSTR + khA];
      a.q[1] = *(const uint4*)&Hs[(wv * 16 + m) * LSTR + khA + 8];
      #pragma unroll
      for (int j = 0; j < 5; ++j) {
        FragU fb;
        fb.q[0] = *(const uint4*)&W2t[(j * 16 + m) * LSTR + khB];
        fb.q[1] = *(const uint4*)&W2t[(j * 16 + m) * LSTR + khB + 8];
        out[j] = wmma_bf16(a.v, fb.v, out[j]);
      }
    }
    __syncthreads();
  }
  // write transposed [B, NM, T] with b2
  #pragma unroll
  for (int j = 0; j < 5; ++j) {
    #pragma unroll
    for (int r = 0; r < 8; ++r) {
      int row = r0 + wv * 16 + mhi + r;
      int b = row >> 11;            // / T (2048)
      int t = row & 2047;
      int col = j * 16 + nlo;
      mel[((long long)b * cNM + col) * cT + t] = out[j][r] + b2[col];
    }
  }
}

// ============================================================================
// Host launch
// ============================================================================
extern "C" void kernel_launch(void* const* d_in, const int* in_sizes, int n_in,
                              void* d_out, int out_size, void* d_ws, size_t ws_size,
                              hipStream_t stream) {
  (void)in_sizes; (void)n_in; (void)out_size; (void)ws_size;
  const int*   text_ids = (const int*)d_in[0];
  const int*   text_len = (const int*)d_in[1];
  // d_in[2] = mel_targets (only target_len=T used, known at compile time)
  const float* emb  = (const float*)d_in[3];
  const float* pe   = (const float*)d_in[4];
  const float* Wqkv = (const float*)d_in[5];
  const float* bqkv = (const float*)d_in[6];
  const float* Wo   = (const float*)d_in[7];
  const float* bo   = (const float*)d_in[8];
  const float* ln1s = (const float*)d_in[9];
  const float* ln1b = (const float*)d_in[10];
  const float* ln2s = (const float*)d_in[11];
  const float* ln2b = (const float*)d_in[12];
  const float* W1   = (const float*)d_in[13];
  const float* b1   = (const float*)d_in[14];
  const float* W2   = (const float*)d_in[15];
  const float* b2   = (const float*)d_in[16];
  const float* mW1  = (const float*)d_in[17];
  const float* mb1  = (const float*)d_in[18];
  const float* mW2  = (const float*)d_in[19];
  const float* mb2  = (const float*)d_in[20];
  const float* dW1  = (const float*)d_in[21];
  const float* db1  = (const float*)d_in[22];
  const float* dW2  = (const float*)d_in[23];
  const float* db2  = (const float*)d_in[24];

  float* out_mel = (float*)d_out;
  float* out_dur = out_mel + (size_t)cB * cNM * cT;
  float* out_enc = out_dur + (size_t)cB * cS;

  char* wp = (char*)d_ws;
  auto alloc = [&](size_t bytes) { char* p = wp; wp += (bytes + 255) & ~(size_t)255; return p; };
  float*          xf    = (float*)         alloc((size_t)cB * cS * cD * 4);
  unsigned short* xb    = (unsigned short*)alloc((size_t)cB * cS * cD * 2);
  float*          qkvf  = (float*)         alloc((size_t)cB * cS * 3 * cD * 4);
  unsigned short* qb    = (unsigned short*)alloc((size_t)cB * cH * cS * cDH * 2);
  unsigned short* kb    = (unsigned short*)alloc((size_t)cB * cH * cS * cDH * 2);
  unsigned short* vbT   = (unsigned short*)alloc((size_t)cB * cH * cDH * cS * 2);
  float*          scf   = (float*)         alloc((size_t)cB * cH * cS * cS * 4);
  unsigned short* attnb = (unsigned short*)alloc((size_t)cB * cH * cS * cS * 2);
  float*          of    = (float*)         alloc((size_t)cB * cH * cS * cDH * 4);
  unsigned short* omb   = (unsigned short*)alloc((size_t)cB * cS * cD * 2);
  float*          pf    = (float*)         alloc((size_t)cB * cS * cD * 4);
  unsigned short* hb    = (unsigned short*)alloc((size_t)cB * cS * cFF * 2);
  unsigned short* durh  = (unsigned short*)alloc((size_t)cB * cS * 256 * 2);
  unsigned short* wbuf  = (unsigned short*)alloc((size_t)cD * cFF * 2);
  unsigned short* w2buf = (unsigned short*)alloc((size_t)cFF * cNM * 2);
  int*            cum   = (int*)           alloc((size_t)cB * cS * 4);
  int*            idxb  = (int*)           alloc((size_t)cB * cT * 4);
  unsigned short* expb  = (unsigned short*)alloc((size_t)cB * cT * cD * 2);

  const int NTOT = cB * cS * cD;     // 2,097,152
  const int MR   = cB * cS;          // 4096 rows

  // transpose-convert: src [K x N] f32 -> dst [N x K] bf16
  auto convT = [&](const float* s, unsigned short* d, int K, int N) {
    dim3 g((N + 31) / 32, (K + 31) / 32);
    k_convT<<<g, 256, 0, stream>>>(s, d, K, N);
  };
  auto gemm = [&](const unsigned short* Ap, const unsigned short* Bp, const float* biasp,
                  void* Cp, int M, int N, int K, long long sA, long long sB, long long sC,
                  int nb, int flags) {
    dim3 g(N / 64, M / 128, nb);
    k_gemm<<<g, 256, 0, stream>>>(Ap, Bp, biasp, Cp, M, N, K, sA, sB, sC, flags);
  };

  k_embed<<<(NTOT + 255) / 256, 256, 0, stream>>>(text_ids, emb, pe, xf, xb);

  for (int l = 0; l < cL; ++l) {
    convT(Wqkv + (size_t)l * cD * 3 * cD, wbuf, cD, 3 * cD);
    gemm(xb, wbuf, bqkv + l * 3 * cD, qkvf, MR, 3 * cD, cD, 0, 0, 0, 1, FL_BIAS);
    k_split_qkv<<<(NTOT + 255) / 256, 256, 0, stream>>>(qkvf, qb, kb, vbT);
    // scores = Q * K^T  (batched over B*H)
    gemm(qb, kb, nullptr, scf, cS, cS, cDH,
         (long long)cS * cDH, (long long)cS * cDH, (long long)cS * cS, cB * cH, 0);
    k_softmax<<<cB * cH * cS, 256, 0, stream>>>(scf, attnb, text_len, 0.125f);
    // o = attn * V = attn * (V^T)^T   (batched over B*H)
    gemm(attnb, vbT, nullptr, of, cS, cDH, cS,
         (long long)cS * cS, (long long)cDH * cS, (long long)cS * cDH, cB * cH, 0);
    k_merge_heads<<<(NTOT + 255) / 256, 256, 0, stream>>>(of, omb);
    convT(Wo + (size_t)l * cD * cD, wbuf, cD, cD);
    gemm(omb, wbuf, bo + l * cD, pf, MR, cD, cD, 0, 0, 0, 1, FL_BIAS);
    k_add_ln<<<MR, 256, 0, stream>>>(xf, pf, ln1s + l * cD, ln1b + l * cD, xb);
    convT(W1 + (size_t)l * cD * cFF, wbuf, cD, cFF);
    gemm(xb, wbuf, b1 + l * cFF, hb, MR, cFF, cD, 0, 0, 0, 1, FL_BIAS | FL_RELU | FL_OBF16);
    convT(W2 + (size_t)l * cFF * cD, wbuf, cFF, cD);
    gemm(hb, wbuf, b2 + l * cD, pf, MR, cD, cFF, 0, 0, 0, 1, FL_BIAS);
    k_add_ln<<<MR, 256, 0, stream>>>(xf, pf, ln2s + l * cD, ln2b + l * cD, xb);
  }

  // enc output
  hipMemcpyAsync(out_enc, xf, (size_t)MR * cD * 4, hipMemcpyDeviceToDevice, stream);

  // duration head
  convT(dW1, wbuf, cD, 256);
  gemm(xb, wbuf, db1, durh, MR, 256, cD, 0, 0, 0, 1, FL_BIAS | FL_RELU | FL_OBF16);
  k_dur2<<<MR, 256, 0, stream>>>(durh, dW2, db2, out_dur);

  // length regulation
  k_cumsum<<<cB, 256, 0, stream>>>(out_dur, cum);
  k_idx<<<(cB * cT + 255) / 256, 256, 0, stream>>>(cum, idxb);
  k_expand<<<(int)(((size_t)cB * cT * cD + 255) / 256), 256, 0, stream>>>(xf, idxb, expb);

  // fused mel decoder (weights pre-transposed: W1T [FF x D], W2T [NM x FF])
  convT(mW1, wbuf, cD, cFF);
  convT(mW2, w2buf, cFF, cNM);
  k_mel<<<cB * cT / 64, 128, 0, stream>>>(expb, wbuf, mb1, w2buf, mb2, out_mel);
}
